// SegmentationBranch_75462575390930
// MI455X (gfx1250) — matrix operand order repaired and checked by reading, not compiled, not run on hardware
//
#include <hip/hip_runtime.h>
#include <hip/hip_bf16.h>

typedef __attribute__((ext_vector_type(16))) __bf16 v16bf;
typedef __attribute__((ext_vector_type(8)))  __bf16 v8bf;
typedef __attribute__((ext_vector_type(8)))  float  v8f;

#define KCH     512     // input channels
#define NOUT    64      // hidden channels
#define KC      128     // K-chunk staged in LDS
#define LWPITCH 136     // row pitch in bf16 (272B: 16B-aligned, lane m -> bank 4m mod 64, conflict-free)

// ---------------------------------------------------------------------------
// conv1x1 + ReLU as a wave-per-16-pixel-tile GEMM using V_WMMA_F32_16X16X32_BF16.
// HW is a template constant so A-loads use immediate offsets off one base
// (chunk-local offset <= 127*4800*4 B, inside the +-8.38MB IOFFSET range).
// ---------------------------------------------------------------------------
template <int HW>
__global__ __launch_bounds__(256) void conv1x1_relu_wmma_bf16(
    const float* __restrict__ feat,   // (B, 512, H, W) fp32
    const float* __restrict__ w,      // (64, 512) fp32
    const float* __restrict__ bias,   // (64,)
    float* __restrict__ out)          // (B*H*W, 64)
{
    __shared__ __bf16 lw[NOUT * LWPITCH];

    const int tid  = threadIdx.x;
    const int lane = tid & 31;
    const int wave = tid >> 5;
    const int m    = lane & 15;           // pixel-in-tile (A) / N-in-tile (B,C,D)
    const int h    = (lane >> 4) & 1;     // lane half select

    const int tileBase = (blockIdx.x * 8 + wave) * 16;
    const int b   = tileBase / HW;        // tiles never cross batch (HW % 16 == 0)
    const int pix = tileBase - b * HW;
    const float* fptr = feat + (b * KCH * HW + pix + m);   // advance by KC*HW per chunk
    const float* wptr = w;                                  // advance by KC per chunk

    v8f acc0 = {}, acc1 = {}, acc2 = {}, acc3 = {};

    for (int chunk = 0; chunk < KCH / KC; ++chunk) {
        __syncthreads();
        // cooperative stage of w[:, chunk*KC : +KC] into LDS as bf16
        // float4 global loads, 4x bf16 (8B) LDS stores
        #pragma unroll
        for (int i = 0; i < (NOUT * KC) / (256 * 4); ++i) {
            int idx4 = (tid + i * 256) * 4;
            int n    = idx4 >> 7;            // KC == 128
            int kc   = idx4 & (KC - 1);
            float4 f = *(const float4*)(wptr + n * KCH + kc);
            __bf16* d = &lw[n * LWPITCH + kc];
            d[0] = (__bf16)f.x; d[1] = (__bf16)f.y;
            d[2] = (__bf16)f.z; d[3] = (__bf16)f.w;
        }
        __syncthreads();

        #pragma unroll
        for (int kk = 0; kk < KC / 32; ++kk) {
            // A fragment: row M = m, contiguous k order shared with B: elem e -> K = h*16 + e
            v16bf a;
            #pragma unroll
            for (int e = 0; e < 16; ++e) {
                int c = kk * 32 + (h << 4) + e;
                a[e] = (__bf16)fptr[c * HW];          // immediate-offset global_load
            }
            // B fragments: contiguous k order shared with A: lane N = m, elem e -> K = h*16 + e
            // 32B contiguous per fragment -> 2x ds_load_b128, conflict-free
            const __bf16* lrow = &lw[kk * 32 + (h << 4)];
            const v8bf* r0 = (const v8bf*)(lrow + ( 0 + m) * LWPITCH);
            const v8bf* r1 = (const v8bf*)(lrow + (16 + m) * LWPITCH);
            const v8bf* r2 = (const v8bf*)(lrow + (32 + m) * LWPITCH);
            const v8bf* r3 = (const v8bf*)(lrow + (48 + m) * LWPITCH);
            v8bf l0 = r0[0], h0 = r0[1];
            v8bf l1 = r1[0], h1 = r1[1];
            v8bf l2 = r2[0], h2 = r2[1];
            v8bf l3 = r3[0], h3 = r3[1];
            v16bf b0 = __builtin_shufflevector(l0, h0, 0,1,2,3,4,5,6,7,8,9,10,11,12,13,14,15);
            v16bf b1 = __builtin_shufflevector(l1, h1, 0,1,2,3,4,5,6,7,8,9,10,11,12,13,14,15);
            v16bf b2 = __builtin_shufflevector(l2, h2, 0,1,2,3,4,5,6,7,8,9,10,11,12,13,14,15);
            v16bf b3 = __builtin_shufflevector(l3, h3, 0,1,2,3,4,5,6,7,8,9,10,11,12,13,14,15);

            acc0 = __builtin_amdgcn_wmma_f32_16x16x32_bf16(false, a, false, b0, (short)0, acc0, false, false);
            acc1 = __builtin_amdgcn_wmma_f32_16x16x32_bf16(false, a, false, b1, (short)0, acc1, false, false);
            acc2 = __builtin_amdgcn_wmma_f32_16x16x32_bf16(false, a, false, b2, (short)0, acc2, false, false);
            acc3 = __builtin_amdgcn_wmma_f32_16x16x32_bf16(false, a, false, b3, (short)0, acc3, false, false);
        }
        fptr += KC * HW;
        wptr += KC;
    }

    // epilogue: D layout — VGPR r holds M = r + 8*h (pixel), N = m. Bias + ReLU.
    float bia0 = bias[ 0 + m];
    float bia1 = bias[16 + m];
    float bia2 = bias[32 + m];
    float bia3 = bias[48 + m];
    #pragma unroll
    for (int r = 0; r < 8; ++r) {
        int px = tileBase + r + 8 * h;
        float* op = out + px * NOUT;
        op[ 0 + m] = fmaxf(acc0[r] + bia0, 0.0f);
        op[16 + m] = fmaxf(acc1[r] + bia1, 0.0f);
        op[32 + m] = fmaxf(acc2[r] + bia2, 0.0f);
        op[48 + m] = fmaxf(acc3[r] + bia3, 0.0f);
    }
}

// ---------------------------------------------------------------------------
// Fused: bilinear upsample f2 (30x40 -> 60x80, align_corners=False) + add f1 +
// 11x64 projection + softmax. prob60 stored channel-major (b,c,60,80).
// ---------------------------------------------------------------------------
__global__ __launch_bounds__(256) void fuse_logits_softmax(
    const float* __restrict__ f1,     // (38400, 64)
    const float* __restrict__ f2,     // (9600, 64)
    const float* __restrict__ wr,     // (11, 64)
    const float* __restrict__ br,     // (11,)
    float* __restrict__ prob60)       // (8, 11, 60, 80)
{
    int p = blockIdx.x * 256 + threadIdx.x;
    if (p >= 8 * 4800) return;
    int b   = p / 4800;
    int rem = p - b * 4800;
    int y   = rem / 80;
    int x   = rem - y * 80;

    float fy = fminf(fmaxf((y + 0.5f) * 0.5f - 0.5f, 0.0f), 29.0f);
    float fx = fminf(fmaxf((x + 0.5f) * 0.5f - 0.5f, 0.0f), 39.0f);
    int y0 = (int)fy; int y1 = min(y0 + 1, 29);
    int x0 = (int)fx; int x1 = min(x0 + 1, 39);
    float wy = fy - (float)y0, wx = fx - (float)x0;

    const float* f2b = f2 + b * 1200 * 64;
    const float* p00 = f2b + (y0 * 40 + x0) * 64;
    const float* p01 = f2b + (y0 * 40 + x1) * 64;
    const float* p10 = f2b + (y1 * 40 + x0) * 64;
    const float* p11 = f2b + (y1 * 40 + x1) * 64;
    const float* f1p = f1 + p * 64;

    float v[64];
    #pragma unroll
    for (int c = 0; c < 64; ++c) {
        float top = p00[c] * (1.0f - wx) + p01[c] * wx;
        float bot = p10[c] * (1.0f - wx) + p11[c] * wx;
        v[c] = top * (1.0f - wy) + bot * wy + f1p[c];
    }

    float logit[11];
    #pragma unroll
    for (int o = 0; o < 11; ++o) {
        float s = br[o];
        const float* wo = wr + o * 64;
        #pragma unroll
        for (int c = 0; c < 64; ++c) s = fmaf(wo[c], v[c], s);
        logit[o] = s;
    }
    float mx = logit[0];
    #pragma unroll
    for (int o = 1; o < 11; ++o) mx = fmaxf(mx, logit[o]);
    float sum = 0.0f;
    #pragma unroll
    for (int o = 0; o < 11; ++o) { logit[o] = __expf(logit[o] - mx); sum += logit[o]; }
    float inv = 1.0f / sum;
    float* pb = prob60 + (b * 11) * 4800 + y * 80 + x;
    #pragma unroll
    for (int o = 0; o < 11; ++o) pb[o * 4800] = logit[o] * inv;
}

// ---------------------------------------------------------------------------
// bbox stat buffers init
// ---------------------------------------------------------------------------
__global__ void init_stats(int* cnt, int* mnx, int* mny, int* mxx, int* mxy)
{
    int t = threadIdx.x;
    if (t < 80) {
        cnt[t] = 0;
        mnx[t] = 1 << 30; mny[t] = 1 << 30;
        mxx[t] = -1;      mxy[t] = -1;
    }
}

// ---------------------------------------------------------------------------
// Final resize (60x80 -> 480x640, align_corners=True) + argmax -> seg +
// per-class bbox stats. 4 consecutive x per thread -> b128 prob/seg stores.
// Blocks never cross a batch (76800 quads per batch / 256 = 300 blocks).
// ---------------------------------------------------------------------------
__global__ __launch_bounds__(256) void resize_argmax_stats(
    const float* __restrict__ prob60,
    float* __restrict__ prob_out,     // (8, 11, 480, 640)
    float* __restrict__ seg_out,      // (8, 480, 640) as float
    int* __restrict__ cnt, int* __restrict__ mnx, int* __restrict__ mny,
    int* __restrict__ mxx, int* __restrict__ mxy)
{
    __shared__ int s_cnt[10], s_mnx[10], s_mny[10], s_mxx[10], s_mxy[10];
    const int tid = threadIdx.x;
    if (tid < 10) {
        s_cnt[tid] = 0;
        s_mnx[tid] = 1 << 30; s_mny[tid] = 1 << 30;
        s_mxx[tid] = -1;      s_mxy[tid] = -1;
    }
    __syncthreads();

    int t   = blockIdx.x * 256 + tid;          // [0, 614400)
    int q0  = t * 4;
    int b   = q0 / 307200;
    int rem = q0 - b * 307200;
    int y   = rem / 640;
    int xb  = rem - y * 640;                   // multiple of 4; quads never cross rows

    float fy = (float)y * (59.0f / 479.0f);
    int y0 = (int)fy; int y1 = min(y0 + 1, 59);
    float wy = fy - (float)y0;

    int   xi0[4], xi1[4];
    float wxv[4];
    #pragma unroll
    for (int j = 0; j < 4; ++j) {
        float fx = (float)(xb + j) * (79.0f / 639.0f);
        int x0 = (int)fx;
        xi0[j] = x0;
        xi1[j] = min(x0 + 1, 79);
        wxv[j] = fx - (float)x0;
    }

    const float* base = prob60 + b * 11 * 4800;
    float best[4] = {-1.0f, -1.0f, -1.0f, -1.0f};
    int   bi[4]   = {0, 0, 0, 0};
    #pragma unroll
    for (int c = 0; c < 11; ++c) {
        const float* py0 = base + c * 4800 + y0 * 80;
        const float* py1 = base + c * 4800 + y1 * 80;
        float4 vv;
        float* vp = (float*)&vv;
        #pragma unroll
        for (int j = 0; j < 4; ++j) {
            float top = py0[xi0[j]] * (1.0f - wxv[j]) + py0[xi1[j]] * wxv[j];
            float bot = py1[xi0[j]] * (1.0f - wxv[j]) + py1[xi1[j]] * wxv[j];
            float v   = top * (1.0f - wy) + bot * wy;
            vp[j] = v;
            if (v > best[j]) { best[j] = v; bi[j] = c; }   // first-max wins (strict >)
        }
        *(float4*)(prob_out + ((b * 11 + c) * 480 + y) * 640 + xb) = vv;
    }
    float4 sv = make_float4((float)bi[0], (float)bi[1], (float)bi[2], (float)bi[3]);
    *(float4*)(seg_out + q0) = sv;

    #pragma unroll
    for (int j = 0; j < 4; ++j) {
        int c = bi[j];
        if (c >= 1 && c <= 9) {
            atomicAdd(&s_cnt[c], 1);
            atomicMin(&s_mnx[c], xb + j); atomicMin(&s_mny[c], y);
            atomicMax(&s_mxx[c], xb + j); atomicMax(&s_mxy[c], y);
        }
    }
    __syncthreads();
    if (tid >= 1 && tid < 10) {
        if (s_cnt[tid] > 0) {
            int g = b * 10 + tid;
            atomicAdd(&cnt[g], s_cnt[tid]);
            atomicMin(&mnx[g], s_mnx[tid]); atomicMin(&mny[g], s_mny[tid]);
            atomicMax(&mxx[g], s_mxx[tid]); atomicMax(&mxy[g], s_mxy[tid]);
        }
    }
}

// ---------------------------------------------------------------------------
// Emit bbx: (8, 10, 6) = [b, minx, miny, maxx, maxy, cls] or -1
// ---------------------------------------------------------------------------
__global__ void make_bbx(const int* __restrict__ cnt, const int* __restrict__ mnx,
                         const int* __restrict__ mny, const int* __restrict__ mxx,
                         const int* __restrict__ mxy, float* __restrict__ bbx)
{
    int t = threadIdx.x;
    if (t >= 80) return;
    int b = t / 10, c = t - b * 10;
    float* o = bbx + t * 6;
    if (c != 0 && cnt[t] >= 500) {
        o[0] = (float)b;
        o[1] = (float)mnx[t]; o[2] = (float)mny[t];
        o[3] = (float)mxx[t]; o[4] = (float)mxy[t];
        o[5] = (float)c;
    } else {
        o[0] = o[1] = o[2] = o[3] = o[4] = o[5] = -1.0f;
    }
}

// ---------------------------------------------------------------------------
extern "C" void kernel_launch(void* const* d_in, const int* in_sizes, int n_in,
                              void* d_out, int out_size, void* d_ws, size_t ws_size,
                              hipStream_t stream)
{
    (void)in_sizes; (void)n_in; (void)out_size; (void)ws_size;

    const float* feature1 = (const float*)d_in[0];  // (8,512,60,80)
    const float* feature2 = (const float*)d_in[1];  // (8,512,30,40)
    const float* w1 = (const float*)d_in[2];        // (64,512)
    const float* b1 = (const float*)d_in[3];
    const float* w2 = (const float*)d_in[4];
    const float* b2 = (const float*)d_in[5];
    const float* wr = (const float*)d_in[6];        // (11,64)
    const float* br = (const float*)d_in[7];

    float* out = (float*)d_out;
    float* prob_out = out;                          // 8*11*480*640 = 27,033,600
    float* seg_out  = out + 27033600;               // 8*480*640   =  2,457,600
    float* bbx_out  = out + 29491200;               // 8*10*6      =        480

    char* ws = (char*)d_ws;
    float* f1     = (float*)(ws);                   //  9,830,400 B
    float* f2     = (float*)(ws + 9830400);         //  2,457,600 B
    float* prob60 = (float*)(ws + 12288000);        //  1,689,600 B
    int*   cnt    = (int*)  (ws + 13977600);
    int*   mnx    = cnt + 80;
    int*   mny    = cnt + 160;
    int*   mxx    = cnt + 240;
    int*   mxy    = cnt + 320;

    // 8 waves/block * 16 px = 128 px per block
    conv1x1_relu_wmma_bf16<4800><<<300, 256, 0, stream>>>(feature1, w1, b1, f1);
    conv1x1_relu_wmma_bf16<1200><<< 75, 256, 0, stream>>>(feature2, w2, b2, f2);
    fuse_logits_softmax        <<<150, 256, 0, stream>>>(f1, f2, wr, br, prob60);
    init_stats                 <<<  1, 128, 0, stream>>>(cnt, mnx, mny, mxx, mxy);
    resize_argmax_stats        <<<2400, 256, 0, stream>>>(prob60, prob_out, seg_out,
                                                          cnt, mnx, mny, mxx, mxy);
    make_bbx                   <<<  1, 128, 0, stream>>>(cnt, mnx, mny, mxx, mxy, bbx_out);
}
